// AttentiveKnrmLayer_60885456388502
// MI455X (gfx1250) — matrix operand order repaired and checked by reading, not compiled, not run on hardware
//
#include <hip/hip_runtime.h>
#include <hip/hip_bf16.h>
#include <math.h>

// ---------------------------------------------------------------------------
// AttentiveKnrmLayer for MI455X (gfx1250, wave32, WMMA).
//  B=64, NB=8, Q=32, T=256, D=256, N_BINS=11.
//  Pipeline:
//   1) rownorm_f16   : L2-normalize hq / can rows, emit f16 (feeds WMMA).
//   2) wmma_sim      : per-batch 256x256x256 GEMM via v_wmma_f32_16x16x32_f16.
//   3) knrm_pool     : Gaussian kernel binning + sum over T + log*0.01*masks.
//   4) finalize      : sum over Q, dense+tanh, attention matvec+softmax, out[B].
// ---------------------------------------------------------------------------

#define DM 256      // D_MODEL, also T, also rows per batch (NB*Q)
#define NBINS 11

__constant__ float c_mus[NBINS] =
    {1.0f, 0.9f, 0.7f, 0.5f, 0.3f, 0.1f, -0.1f, -0.3f, -0.5f, -0.7f, -0.9f};
// 1/(2*sigma^2): sigma0=1e-3 -> 5e5 ; sigma=0.1 -> 50
__constant__ float c_inv2s2[NBINS] =
    {5.0e5f, 50.f, 50.f, 50.f, 50.f, 50.f, 50.f, 50.f, 50.f, 50.f, 50.f};

typedef _Float16 v16h __attribute__((ext_vector_type(16)));
typedef _Float16 v8h  __attribute__((ext_vector_type(8)));
typedef float    v8f  __attribute__((ext_vector_type(8)));

// ---------------------------------------------------------------------------
// Kernel 1: L2-normalize each 256-element row, store as f16.
// grid = nrows, block = 256 (8 waves).
// ---------------------------------------------------------------------------
__global__ __launch_bounds__(256) void rownorm_f16(
    const float* __restrict__ src, _Float16* __restrict__ dst) {
  const int row  = blockIdx.x;
  const int t    = threadIdx.x;
  const int lane = t & 31;
  const int w    = t >> 5;

  float x  = src[(size_t)row * DM + t];
  float ss = x * x;
  #pragma unroll
  for (int m = 16; m > 0; m >>= 1) ss += __shfl_xor(ss, m, 32);

  __shared__ float wsum[8];
  __shared__ float inv_s;
  if (lane == 0) wsum[w] = ss;
  __syncthreads();
  if (t == 0) {
    float s = 0.f;
    #pragma unroll
    for (int i = 0; i < 8; ++i) s += wsum[i];
    inv_s = 1.0f / fmaxf(sqrtf(s), 1e-10f);
  }
  __syncthreads();
  dst[(size_t)row * DM + t] = (_Float16)(x * inv_s);
}

// ---------------------------------------------------------------------------
// Kernel 2: per-batch GEMM  M[b] = A[b] (256x256) * B[b]^T (256x256)
// via v_wmma_f32_16x16x32_f16. One wave per (batch, 16-row tile); A frags
// are preloaded once and reused across all 16 N tiles -> 128 WMMAs/wave.
// grid = 64*16 blocks, block = 32 (one wave, EXEC all ones).
// ---------------------------------------------------------------------------
__global__ __launch_bounds__(32) void wmma_sim(
    const _Float16* __restrict__ A, const _Float16* __restrict__ Bm,
    float* __restrict__ M) {
  const int b     = blockIdx.x >> 4;
  const int rtile = blockIdx.x & 15;
  const int lane  = threadIdx.x;   // 0..31
  const int half  = lane >> 4;     // K-stripe select
  const int l15   = lane & 15;     // row (A) / col (B,C)

  const _Float16* Ab = A  + (size_t)b * DM * DM;
  const _Float16* Bb = Bm + (size_t)b * DM * DM;
  float*          Mb = M  + (size_t)b * DM * DM;

  // --- preload 8 A fragments (K = ks*32 .. +31), contiguous-k A layout (same order as B):
  //     lane l15 = row; elems 0..7  -> K = kb + 16*half + 0..7
  //                     elems 8..15 -> K = kb + 16*half + 8..15
  v16h afrag[8];
  const _Float16* arow = Ab + (size_t)(rtile * 16 + l15) * DM;
  #pragma unroll
  for (int ks = 0; ks < 8; ++ks) {
    const _Float16* p = arow + ks * 32 + half * 16;
    v8h lo = *(const v8h*)(p);
    v8h hi = *(const v8h*)(p + 8);
    afrag[ks] = __builtin_shufflevector(lo, hi,
        0, 1, 2, 3, 4, 5, 6, 7, 8, 9, 10, 11, 12, 13, 14, 15);
  }

  // --- sweep the 16 N tiles; B frag per K-step is one contiguous 32B load:
  //     lane col = l15, elem i -> K = kb + 16*half + i
  for (int nt = 0; nt < 16; ++nt) {
    v8f c = {};
    const _Float16* brow = Bb + (size_t)(nt * 16 + l15) * DM + half * 16;
    #pragma unroll
    for (int ks = 0; ks < 8; ++ks) {
      v16h bfrag = *(const v16h*)(brow + ks * 32);
      c = __builtin_amdgcn_wmma_f32_16x16x32_f16(
          /*neg_a=*/false, afrag[ks], /*neg_b=*/false, bfrag,
          /*c_mod=*/(short)0, c, /*reuse_a=*/false, /*reuse_b=*/false);
    }
    // C layout: VGPR r -> row = rtile*16 + r + 8*half, col = nt*16 + l15
    float* mout = Mb + (size_t)(rtile * 16 + half * 8) * DM + nt * 16 + l15;
    #pragma unroll
    for (int r = 0; r < 8; ++r) mout[(size_t)r * DM] = c[r];
  }
}

// ---------------------------------------------------------------------------
// Kernel 3: Gaussian soft-binning + pooling over T, then log*0.01*masks.
// grid = B*256 rows (row = b*256 + nb*32 + q), block = 256 (one thread per T).
// ---------------------------------------------------------------------------
__global__ __launch_bounds__(256) void knrm_pool(
    const float* __restrict__ M, const float* __restrict__ mask_can,
    const float* __restrict__ mask_hq, const float* __restrict__ word_atten,
    float* __restrict__ lp) {
  const int row  = blockIdx.x;
  const int b    = row >> 8;
  const int t    = threadIdx.x;
  const int lane = t & 31;
  const int w    = t >> 5;

  const float m  = M[(size_t)row * DM + t];
  const float km = mask_can[b * DM + t];

  float acc[NBINS];
  #pragma unroll
  for (int j = 0; j < NBINS; ++j) {
    float d = m - c_mus[j];
    acc[j] = __expf(-d * d * c_inv2s2[j]) * km;
  }
  #pragma unroll
  for (int j = 0; j < NBINS; ++j) {
    #pragma unroll
    for (int s = 16; s > 0; s >>= 1) acc[j] += __shfl_xor(acc[j], s, 32);
  }

  __shared__ float ws[8][NBINS];
  if (lane == 0) {
    #pragma unroll
    for (int j = 0; j < NBINS; ++j) ws[w][j] = acc[j];
  }
  __syncthreads();

  if (t < NBINS) {
    float s = 0.f;
    #pragma unroll
    for (int w2 = 0; w2 < 8; ++w2) s += ws[w2][t];
    float l = __logf(fmaxf(s, 1e-10f)) * 0.01f;
    l = l * mask_hq[row] * word_atten[row];
    lp[(size_t)row * NBINS + t] = l;
  }
}

// ---------------------------------------------------------------------------
// Kernel 4: finalize per batch. block = 256 (8 waves, wave w handles nb=w).
//  - q = rep_cur @ W_att + b_att (coalesced column matvec into LDS)
//  - attn logits = <rep[b,nb], q>/16, masked softmax over 8 blocks
//  - score = tanh(mask_session * (sum_q lp) @ W_dense + b_dense)
//  - out[b] = sum score*attn
// ---------------------------------------------------------------------------
__global__ __launch_bounds__(256) void finalize(
    const float* __restrict__ lp, const float* __restrict__ rep,
    const float* __restrict__ rep_cur, const float* __restrict__ mask_session,
    const float* __restrict__ W_dense, const float* __restrict__ b_dense,
    const float* __restrict__ W_att, const float* __restrict__ b_att,
    float* __restrict__ out) {
  const int b    = blockIdx.x;
  const int t    = threadIdx.x;
  const int lane = t & 31;
  const int w    = t >> 5;

  __shared__ float qv[DM];
  __shared__ float logit[8];
  __shared__ float score[8];

  {   // q_vec[t] = b_att[t] + sum_d rep_cur[d] * W_att[d, t]
    float acc = b_att[t];
    const float* rc = rep_cur + (size_t)b * DM;
    for (int d = 0; d < DM; ++d) acc += rc[d] * W_att[(size_t)d * DM + t];
    qv[t] = acc;
  }
  __syncthreads();

  {   // wave w -> session block nb = w
    const float ms = mask_session[b * 8 + w];

    // attention logit
    const float* rrow = rep + ((size_t)b * 8 + w) * DM;
    float s = 0.f;
    #pragma unroll
    for (int i = 0; i < 8; ++i) { int d = lane + 32 * i; s += rrow[d] * qv[d]; }
    #pragma unroll
    for (int m = 16; m > 0; m >>= 1) s += __shfl_xor(s, m, 32);
    if (lane == 0) logit[w] = (ms > 0.f) ? (s * 0.0625f) : -1e9f;

    // pooled @ W_dense  (lane = q position, 32 q's per block)
    float p = 0.f;
    const float* lrow = lp + ((size_t)b * DM + w * 32 + lane) * NBINS;
    #pragma unroll
    for (int j = 0; j < NBINS; ++j) p += lrow[j] * W_dense[j];
    #pragma unroll
    for (int m = 16; m > 0; m >>= 1) p += __shfl_xor(p, m, 32);
    if (lane == 0) score[w] = tanhf(ms * p + b_dense[0]);
  }
  __syncthreads();

  if (t == 0) {
    float mx = -1e30f;
    #pragma unroll
    for (int n = 0; n < 8; ++n) mx = fmaxf(mx, logit[n]);
    float den = 0.f, num = 0.f;
    #pragma unroll
    for (int n = 0; n < 8; ++n) {
      float e = __expf(logit[n] - mx);
      den += e;
      num += e * score[n];
    }
    out[b] = num / den;
  }
}

// ---------------------------------------------------------------------------
extern "C" void kernel_launch(void* const* d_in, const int* in_sizes, int n_in,
                              void* d_out, int out_size, void* d_ws, size_t ws_size,
                              hipStream_t stream) {
  (void)in_sizes; (void)n_in; (void)out_size; (void)ws_size;

  const float* word_atten   = (const float*)d_in[0];   // [64,8,32]
  const float* hq           = (const float*)d_in[1];   // [64,8,32,256]
  const float* can          = (const float*)d_in[2];   // [64,256,256]
  const float* rep          = (const float*)d_in[3];   // [64,8,256]
  const float* rep_cur      = (const float*)d_in[4];   // [64,256]
  const float* mask_hq      = (const float*)d_in[5];   // [64,8,32]
  const float* mask_can     = (const float*)d_in[6];   // [64,256]
  const float* mask_session = (const float*)d_in[7];   // [64,8]
  const float* W_dense      = (const float*)d_in[8];   // [11,1]
  const float* b_dense      = (const float*)d_in[9];   // [1]
  const float* W_att        = (const float*)d_in[10];  // [256,256]
  const float* b_att        = (const float*)d_in[11];  // [256]
  float* out = (float*)d_out;                          // [64]

  char* ws = (char*)d_ws;
  _Float16* hq_h   = (_Float16*)(ws);                    //  8 MB: [64][256][256] f16
  _Float16* can_h  = (_Float16*)(ws + (size_t)( 8u<<20)); //  8 MB: [64][256][256] f16
  float*    matrix = (float*)   (ws + (size_t)(16u<<20)); // 16 MB: [64][256][256] f32
  float*    lp     = (float*)   (ws + (size_t)(32u<<20)); // 720 KB: [64*256][11] f32

  const int nrows = 64 * 256;   // 16384 rows of 256 for both hq and can

  rownorm_f16<<<nrows, 256, 0, stream>>>(hq,  hq_h);
  rownorm_f16<<<nrows, 256, 0, stream>>>(can, can_h);
  wmma_sim   <<<64 * 16, 32, 0, stream>>>(hq_h, can_h, matrix);
  knrm_pool  <<<nrows, 256, 0, stream>>>(matrix, mask_can, mask_hq, word_atten, lp);
  finalize   <<<64, 256, 0, stream>>>(lp, rep, rep_cur, mask_session,
                                      W_dense, b_dense, W_att, b_att, out);
}
